// PointNetPP_18691697672886
// MI455X (gfx1250) — matrix operand order repaired and checked by reading, not compiled, not run on hardware
//
#include <hip/hip_runtime.h>
#include <hip/hip_bf16.h>

// ---------------------------------------------------------------------------
// PointNet++ (SA x3 + FP x3) for gfx1250. GEMMs use v_wmma_f32_16x16x32_f16
// with fragment-layout weights (coalesced b128 loads) and unrolled K loops.
// ---------------------------------------------------------------------------

typedef _Float16 v16h __attribute__((ext_vector_type(16)));
typedef _Float16 v8h  __attribute__((ext_vector_type(8)));
typedef float    v8f  __attribute__((ext_vector_type(8)));
typedef _Float16 h16;

#define N0 8192
#define KNBR 32
#define R2 4.0f

// ---------------- activation padding (f32 -> padded f16, row-major) --------
__global__ void pnpp_pad_a(const float* __restrict__ in, h16* __restrict__ out,
                           int K, int N, int Np, long total) {
  long e = (long)blockIdx.x * blockDim.x + threadIdx.x;
  if (e >= total) return;
  long r = e / Np; int c = (int)(e % Np);
  out[e] = (r < K && c < N) ? (h16)in[r * (long)N + c] : (h16)0.0f;
}

// ---------------- weight padding into WMMA B-fragment layout ---------------
// out[((kt*ntiles + nt)*32 + lane)*16 + i] = W[kt*32 + (lane>>4)*8 + (i&7) + (i>>3)*16][nt*16 + (lane&15)]
__global__ void pnpp_pad_wfrag(const float* __restrict__ in, h16* __restrict__ out,
                               int K, int N, int ntiles, long total) {
  long e = (long)blockIdx.x * blockDim.x + threadIdx.x;
  if (e >= total) return;
  int i    = (int)(e & 15);
  int lane = (int)((e >> 4) & 31);
  long ft  = e >> 9;
  int nt   = (int)(ft % ntiles);
  int kt   = (int)(ft / ntiles);
  int krow = kt * 32 + ((lane >> 4) << 3) + (i & 7) + ((i >> 3) << 4);
  int col  = (nt << 4) + (lane & 15);
  out[e] = (krow < K && col < N) ? (h16)in[(long)krow * N + col] : (h16)0.0f;
}

__global__ void pnpp_pad_b(const float* __restrict__ in, float* __restrict__ out,
                           int N, int Np) {
  int i = blockIdx.x * blockDim.x + threadIdx.x;
  if (i >= Np) return;
  out[i] = (i < N) ? in[i] : 0.0f;
}

// ---------------- WMMA GEMM: C = act(A[M,KP] @ W[KP,Np] + b) ---------------
// One wave computes a 16 x (16*NT) strip; A fragment reused across NT tiles.
// W is pre-swizzled into per-lane-contiguous fragments -> 2x b128 per tile.
template <int KP, int NT, bool OUTF32>
__global__ __launch_bounds__(256) void pnpp_gemm_wmma(
    const h16* __restrict__ A, int lda,
    const h16* __restrict__ Wfrag,
    const float* __restrict__ bias,
    int M, int Np,
    h16* __restrict__ outH, float* __restrict__ outF,
    int ldc, int nValid, int relu) {
  int lane = threadIdx.x & 31;
  int wave = threadIdx.x >> 5;
  int ntiles  = Np >> 4;
  int ngroups = ntiles / NT;
  int mtiles  = M >> 4;
  int g = blockIdx.x * 8 + wave;
  if (g >= mtiles * ngroups) return;               // wave-uniform exit
  int mt  = g / ngroups;
  int nt0 = (g % ngroups) * NT;
  int row0 = mt << 4;
  int rlo = lane & 15, hi = lane >> 4;

  v8f acc[NT];
#pragma unroll
  for (int t = 0; t < NT; t++) acc[t] = (v8f){};

#pragma unroll
  for (int kt = 0; kt < KP / 32; kt++) {
    const int k0 = kt * 32;
    // A fragment: lane<16 -> row=lane, K = {k0+hi*8..+7} U {k0+16+hi*8..+7}
    const h16* ap = A + (size_t)(row0 + rlo) * lda + k0 + hi * 8;
    v8h alo = *(const v8h*)(ap);
    v8h ahi = *(const v8h*)(ap + 16);
    v16h a = __builtin_shufflevector(alo, ahi, 0, 1, 2, 3, 4, 5, 6, 7,
                                     8, 9, 10, 11, 12, 13, 14, 15);
#pragma unroll
    for (int t = 0; t < NT; t++) {
      const h16* bp = Wfrag + (((size_t)kt * ntiles + (nt0 + t)) * 32 + lane) * 16;
      v16h b = *(const v16h*)bp;
      acc[t] = __builtin_amdgcn_wmma_f32_16x16x32_f16(false, a, false, b,
                                                      (short)0, acc[t], false, false);
    }
  }

#pragma unroll
  for (int t = 0; t < NT; t++) {
    int col = ((nt0 + t) << 4) + rlo;
    float bv = bias[col];
#pragma unroll
    for (int r = 0; r < 8; r++) {
      int row = row0 + r + (hi << 3);
      float v = acc[t][r] + bv;
      if (relu) v = fmaxf(v, 0.0f);
      if (OUTF32) {
        if (col < nValid) outF[(size_t)row * ldc + col] = v;
      } else {
        outH[(size_t)row * ldc + col] = (h16)v;
      }
    }
  }
}

// ---------------- farthest point sampling (serial chain, 1 workgroup) ------
__global__ __launch_bounds__(1024) void pnpp_fps(const float* __restrict__ pos,
                                                 int n, int m,
                                                 int* __restrict__ idx_out) {
  __shared__ float sv[1024];
  __shared__ int   si[1024];
  __shared__ float lp[3];
  __shared__ int   slast;
  int t = threadIdx.x;
  int npt = n >> 10;                       // 8 / 4 / 2 points per thread
  float mind[8];
#pragma unroll
  for (int k = 0; k < 8; k++) mind[k] = 1e30f;
  if (t == 0) { idx_out[0] = 0; slast = 0; }
  __syncthreads();
  for (int s = 1; s < m; s++) {
    if (t < 3) lp[t] = pos[3 * slast + t];
    __syncthreads();
    float bv = -1e30f; int bi = 0x7fffffff;
    for (int k = 0; k < npt; k++) {
      int i = t + (k << 10);
      float dx = pos[3 * i] - lp[0];
      float dy = pos[3 * i + 1] - lp[1];
      float dz = pos[3 * i + 2] - lp[2];
      float d = dx * dx + dy * dy + dz * dz;
      float md = fminf(mind[k], d);
      mind[k] = md;
      if (md > bv || (md == bv && i < bi)) { bv = md; bi = i; }
    }
    sv[t] = bv; si[t] = bi;
    __syncthreads();
    for (int off = 512; off > 0; off >>= 1) {
      if (t < off) {
        float ov = sv[t + off]; int oi = si[t + off];
        if (ov > sv[t] || (ov == sv[t] && oi < si[t])) { sv[t] = ov; si[t] = oi; }
      }
      __syncthreads();
    }
    if (t == 0) { slast = si[0]; idx_out[s] = si[0]; }
    __syncthreads();
  }
}

__global__ void pnpp_gather_pos(const float* __restrict__ pl,
                                const int* __restrict__ idx, int m,
                                float* __restrict__ pout) {
  int e = blockIdx.x * blockDim.x + threadIdx.x;
  if (e >= m * 3) return;
  pout[e] = pl[idx[e / 3] * 3 + (e % 3)];
}

// ---------------- 32-NN (stable, lexicographic (d2, idx)) ------------------
__global__ __launch_bounds__(256) void pnpp_knn32(const float* __restrict__ pos,
                                                  int n,
                                                  const float* __restrict__ q,
                                                  int* __restrict__ cols,
                                                  int* __restrict__ maskb) {
  int c = blockIdx.x, t = threadIdx.x;
  float qx = q[3 * c], qy = q[3 * c + 1], qz = q[3 * c + 2];
  int npt = n >> 8;                         // up to 32 points/thread
  float d2l[32];
  for (int k = 0; k < npt; k++) {
    int i = t + (k << 8);
    float dx = pos[3 * i] - qx, dy = pos[3 * i + 1] - qy, dz = pos[3 * i + 2] - qz;
    d2l[k] = dx * dx + dy * dy + dz * dz;
  }
  __shared__ float sv[256];
  __shared__ int   si[256];
  __shared__ float lastD;
  __shared__ int   lastI;
  if (t == 0) { lastD = -1e30f; lastI = -1; }
  __syncthreads();
  for (int r = 0; r < KNBR; r++) {
    float ld = lastD; int li = lastI;
    float bv = 3.4e38f; int bi = 0x7fffffff;
    for (int k = 0; k < npt; k++) {
      int i = t + (k << 8);
      float d = d2l[k];
      bool valid = (d > ld) || (d == ld && i > li);
      if (valid && (d < bv || (d == bv && i < bi))) { bv = d; bi = i; }
    }
    sv[t] = bv; si[t] = bi;
    __syncthreads();
    for (int off = 128; off > 0; off >>= 1) {
      if (t < off) {
        float ov = sv[t + off]; int oi = si[t + off];
        if (ov < sv[t] || (ov == sv[t] && oi < si[t])) { sv[t] = ov; si[t] = oi; }
      }
      __syncthreads();
    }
    if (t == 0) {
      cols[(size_t)c * KNBR + r] = si[0];
      maskb[(size_t)c * KNBR + r] = (sv[0] <= R2) ? 1 : 0;
      lastD = sv[0]; lastI = si[0];
    }
    __syncthreads();
  }
}

// ---------------- build SA conv input: [M*32, 160] = cat(x_j, pos_j - q_i) -
__global__ void pnpp_build_sa(const h16* __restrict__ feat,
                              const float* __restrict__ pos,
                              const float* __restrict__ q,
                              const int* __restrict__ cols,
                              long total, h16* __restrict__ A) {
  long e = (long)blockIdx.x * blockDim.x + threadIdx.x;
  if (e >= total) return;
  int j = (int)(e % 160);
  long r = e / 160;
  long c = r / KNBR;
  int col = cols[r];
  h16 v;
  if (j < 128) v = feat[(size_t)col * 128 + j];
  else if (j < 131) {
    int d = j - 128;
    v = (h16)(pos[3 * col + d] - q[3 * c + d]);
  } else v = (h16)0.0f;
  A[e] = v;
}

// ---------------- masked max aggregation over 32 neighbors -----------------
__global__ __launch_bounds__(128) void pnpp_maxagg(const h16* __restrict__ h,
                                                   const int* __restrict__ maskb,
                                                   h16* __restrict__ out) {
  int c = blockIdx.x, ch = threadIdx.x;
  float m = -1e30f; int any = 0;
  for (int k = 0; k < KNBR; k++) {
    if (maskb[(size_t)c * KNBR + k]) {
      any = 1;
      m = fmaxf(m, (float)h[((size_t)c * KNBR + k) * 128 + ch]);
    }
  }
  out[(size_t)c * 128 + ch] = (h16)(any ? m : 0.0f);
}

// ---------------- 3-NN + inverse-d2 weights --------------------------------
__global__ void pnpp_knn3(const float* __restrict__ pc, int nc,
                          const float* __restrict__ pf, int nf,
                          int* __restrict__ ki, float* __restrict__ kw,
                          float* __restrict__ kws) {
  int i = blockIdx.x * blockDim.x + threadIdx.x;
  if (i >= nf) return;
  float x = pf[3 * i], y = pf[3 * i + 1], z = pf[3 * i + 2];
  float bd0 = 3.4e38f, bd1 = 3.4e38f, bd2 = 3.4e38f;
  int bi0 = -1, bi1 = -1, bi2 = -1;
  for (int j = 0; j < nc; j++) {
    float dx = pc[3 * j] - x, dy = pc[3 * j + 1] - y, dz = pc[3 * j + 2] - z;
    float d = dx * dx + dy * dy + dz * dz;
    if (d < bd0 || (d == bd0 && j < bi0)) {
      bd2 = bd1; bi2 = bi1; bd1 = bd0; bi1 = bi0; bd0 = d; bi0 = j;
    } else if (d < bd1 || (d == bd1 && j < bi1)) {
      bd2 = bd1; bi2 = bi1; bd1 = d; bi1 = j;
    } else if (d < bd2 || (d == bd2 && j < bi2)) {
      bd2 = d; bi2 = j;
    }
  }
  float w0 = 1.0f / fmaxf(bd0, 1e-16f);
  float w1 = 1.0f / fmaxf(bd1, 1e-16f);
  float w2 = 1.0f / fmaxf(bd2, 1e-16f);
  ki[i * 3 + 0] = bi0; ki[i * 3 + 1] = bi1; ki[i * 3 + 2] = bi2;
  kw[i * 3 + 0] = w0;  kw[i * 3 + 1] = w1;  kw[i * 3 + 2] = w2;
  kws[i] = w0 + w1 + w2;
}

// ---------------- interpolate + concat skip -> [Nf, 256] -------------------
__global__ __launch_bounds__(128) void pnpp_interp_concat(
    const h16* __restrict__ fc, const int* __restrict__ ki,
    const float* __restrict__ kw, const float* __restrict__ kws,
    const h16* __restrict__ skip, h16* __restrict__ out) {
  int i = blockIdx.x, ch = threadIdx.x;
  float acc = 0.0f;
#pragma unroll
  for (int k = 0; k < 3; k++)
    acc += kw[i * 3 + k] * (float)fc[(size_t)ki[i * 3 + k] * 128 + ch];
  out[(size_t)i * 256 + ch] = (h16)(acc / kws[i]);
  out[(size_t)i * 256 + 128 + ch] = skip[(size_t)i * 128 + ch];
}

// ===========================================================================
extern "C" void kernel_launch(void* const* d_in, const int* in_sizes, int n_in,
                              void* d_out, int out_size, void* d_ws, size_t ws_size,
                              hipStream_t stream) {
  (void)in_sizes; (void)n_in; (void)out_size; (void)ws_size;
  const float* x       = (const float*)d_in[0];
  const float* pos     = (const float*)d_in[1];
  const float* li_w1   = (const float*)d_in[2];
  const float* li_b1   = (const float*)d_in[3];
  const float* li_w2   = (const float*)d_in[4];
  const float* li_b2   = (const float*)d_in[5];
  const float* sa_w1   = (const float*)d_in[6];
  const float* sa_b1   = (const float*)d_in[7];
  const float* sa_w2   = (const float*)d_in[8];
  const float* sa_b2   = (const float*)d_in[9];
  const float* fp_w1   = (const float*)d_in[10];
  const float* fp_b1   = (const float*)d_in[11];
  const float* fp_w2   = (const float*)d_in[12];
  const float* fp_b2   = (const float*)d_in[13];
  const float* lo_w1   = (const float*)d_in[14];
  const float* lo_b1   = (const float*)d_in[15];
  const float* lo_w2   = (const float*)d_in[16];
  const float* lo_b2   = (const float*)d_in[17];
  float* out = (float*)d_out;

  // ---- workspace layout --------------------------------------------------
  char* base = (char*)d_ws;
  size_t off = 0;
  auto alloc = [&](size_t bytes) -> char* {
    off = (off + 255) & ~(size_t)255;
    char* p = base + off;
    off += bytes;
    return p;
  };
  h16*   X16    = (h16*)alloc((size_t)N0 * 32 * 2);
  h16*   wLi1   = (h16*)alloc(32 * 128 * 2);
  h16*   wLi2   = (h16*)alloc(128 * 128 * 2);
  h16*   wSa1[3]; h16* wSa2[3]; h16* wFp1[3]; h16* wFp2[3];
  for (int l = 0; l < 3; l++) wSa1[l] = (h16*)alloc(160 * 160 * 2);
  for (int l = 0; l < 3; l++) wSa2[l] = (h16*)alloc(160 * 128 * 2);
  for (int l = 0; l < 3; l++) wFp1[l] = (h16*)alloc(256 * 256 * 2);
  for (int l = 0; l < 3; l++) wFp2[l] = (h16*)alloc(256 * 128 * 2);
  h16*   wLo1   = (h16*)alloc(128 * 128 * 2);
  h16*   wLo2   = (h16*)alloc(128 * 16 * 2);
  float* bLi1   = (float*)alloc(128 * 4);
  float* bLi2   = (float*)alloc(128 * 4);
  float* bSa1[3]; float* bSa2[3]; float* bFp1[3]; float* bFp2[3];
  for (int l = 0; l < 3; l++) bSa1[l] = (float*)alloc(160 * 4);
  for (int l = 0; l < 3; l++) bSa2[l] = (float*)alloc(128 * 4);
  for (int l = 0; l < 3; l++) bFp1[l] = (float*)alloc(256 * 4);
  for (int l = 0; l < 3; l++) bFp2[l] = (float*)alloc(128 * 4);
  float* bLo1   = (float*)alloc(128 * 4);
  float* bLo2   = (float*)alloc(16 * 4);

  h16*   H1     = (h16*)alloc((size_t)N0 * 128 * 2);
  h16*   feat[4];
  feat[0] = (h16*)alloc((size_t)8192 * 128 * 2);
  feat[1] = (h16*)alloc((size_t)4096 * 128 * 2);
  feat[2] = (h16*)alloc((size_t)2048 * 128 * 2);
  feat[3] = (h16*)alloc((size_t)1024 * 128 * 2);
  float* p1 = (float*)alloc(4096 * 3 * 4);
  float* p2 = (float*)alloc(2048 * 3 * 4);
  float* p3 = (float*)alloc(1024 * 3 * 4);
  int* idx1 = (int*)alloc(4096 * 4);
  int* idx2 = (int*)alloc(2048 * 4);
  int* idx3 = (int*)alloc(1024 * 4);
  int* cols  = (int*)alloc((size_t)4096 * KNBR * 4);
  int* maskb = (int*)alloc((size_t)4096 * KNBR * 4);
  h16* Asa = (h16*)alloc((size_t)4096 * KNBR * 160 * 2);   // reused as Hc2
  h16* Hc1 = (h16*)alloc((size_t)4096 * KNBR * 160 * 2);
  h16* Hc2 = Asa;                                          // alias (Asa dead after GEMM1)
  int*   ki3 = (int*)alloc((size_t)N0 * 3 * 4);
  float* kw3 = (float*)alloc((size_t)N0 * 3 * 4);
  float* kws = (float*)alloc((size_t)N0 * 4);
  h16* Ycat = (h16*)alloc((size_t)N0 * 256 * 2);
  h16* Tfp  = (h16*)alloc((size_t)N0 * 256 * 2);
  h16* xfA  = (h16*)alloc((size_t)N0 * 128 * 2);
  h16* xfB  = (h16*)alloc((size_t)N0 * 128 * 2);
  h16* Lout = (h16*)alloc((size_t)N0 * 128 * 2);

  auto pada = [&](const float* in, h16* o, int K, int N, int Np) {
    long total = (long)K * Np;
    pnpp_pad_a<<<(int)((total + 255) / 256), 256, 0, stream>>>(in, o, K, N, Np, total);
  };
  auto padw = [&](const float* in, h16* o, int K, int N, int Kp, int Np) {
    long total = (long)Kp * Np;   // = ktiles*ntiles*512
    pnpp_pad_wfrag<<<(int)((total + 255) / 256), 256, 0, stream>>>(
        in, o, K, N, Np >> 4, total);
  };
  auto padb = [&](const float* in, float* o, int N, int Np) {
    pnpp_pad_b<<<(Np + 127) / 128, 128, 0, stream>>>(in, o, N, Np);
  };

#define PNPP_GEMM(KP, NT, A_, LDA, W_, B_, M_, NP, O_)                        \
  do {                                                                        \
    int groups_ = ((M_) >> 4) * (((NP) >> 4) / (NT));                         \
    pnpp_gemm_wmma<KP, NT, false><<<(groups_ + 7) / 8, 256, 0, stream>>>(     \
        A_, LDA, W_, B_, M_, NP, O_, nullptr, NP, NP, 1);                     \
  } while (0)

  // ---- weight prep -------------------------------------------------------
  pada(x, X16, N0, 16, 32);
  padw(li_w1, wLi1, 16, 128, 32, 128);   padb(li_b1, bLi1, 128, 128);
  padw(li_w2, wLi2, 128, 128, 128, 128); padb(li_b2, bLi2, 128, 128);
  for (int l = 0; l < 3; l++) {
    padw(sa_w1 + (size_t)l * 131 * 131, wSa1[l], 131, 131, 160, 160);
    padb(sa_b1 + (size_t)l * 131, bSa1[l], 131, 160);
    padw(sa_w2 + (size_t)l * 131 * 128, wSa2[l], 131, 128, 160, 128);
    padb(sa_b2 + (size_t)l * 128, bSa2[l], 128, 128);
    padw(fp_w1 + (size_t)l * 256 * 256, wFp1[l], 256, 256, 256, 256);
    padb(fp_b1 + (size_t)l * 256, bFp1[l], 256, 256);
    padw(fp_w2 + (size_t)l * 256 * 128, wFp2[l], 256, 128, 256, 128);
    padb(fp_b2 + (size_t)l * 128, bFp2[l], 128, 128);
  }
  padw(lo_w1, wLo1, 128, 128, 128, 128); padb(lo_b1, bLo1, 128, 128);
  padw(lo_w2, wLo2, 128, 2, 128, 16);    padb(lo_b2, bLo2, 2, 16);

  // ---- lin_in ------------------------------------------------------------
  PNPP_GEMM(32, 2, X16, 32, wLi1, bLi1, N0, 128, H1);
  PNPP_GEMM(128, 2, H1, 128, wLi2, bLi2, N0, 128, feat[0]);

  // ---- SA levels ---------------------------------------------------------
  const int Ns[4] = {8192, 4096, 2048, 1024};
  const float* pbuf[4] = {pos, p1, p2, p3};
  float* pout[4] = {nullptr, p1, p2, p3};
  int* idxb[3] = {idx1, idx2, idx3};
  for (int l = 0; l < 3; l++) {
    int n = Ns[l], m = Ns[l + 1];
    pnpp_fps<<<1, 1024, 0, stream>>>(pbuf[l], n, m, idxb[l]);
    pnpp_gather_pos<<<(m * 3 + 255) / 256, 256, 0, stream>>>(pbuf[l], idxb[l], m, pout[l + 1]);
    pnpp_knn32<<<m, 256, 0, stream>>>(pbuf[l], n, pbuf[l + 1], cols, maskb);
    long total = (long)m * KNBR * 160;
    pnpp_build_sa<<<(int)((total + 255) / 256), 256, 0, stream>>>(
        feat[l], pbuf[l], pbuf[l + 1], cols, total, Asa);
    PNPP_GEMM(160, 2, Asa, 160, wSa1[l], bSa1[l], m * KNBR, 160, Hc1);
    PNPP_GEMM(160, 2, Hc1, 160, wSa2[l], bSa2[l], m * KNBR, 128, Hc2);
    pnpp_maxagg<<<m, 128, 0, stream>>>(Hc2, maskb, feat[l + 1]);
  }

  // ---- FP levels (deepest first) ----------------------------------------
  const h16* cf = feat[3];
  h16* xf_bufs[3] = {xfA, xfB, xfA};   // j=2 -> xfA, j=1 -> xfB, j=0 -> xfA
  for (int i = 0; i < 3; i++) {
    int j = 2 - i;
    int nc = Ns[j + 1], nf = Ns[j];
    pnpp_knn3<<<(nf + 127) / 128, 128, 0, stream>>>(pbuf[j + 1], nc, pbuf[j], nf,
                                                    ki3, kw3, kws);
    pnpp_interp_concat<<<nf, 128, 0, stream>>>(cf, ki3, kw3, kws, feat[j], Ycat);
    PNPP_GEMM(256, 2, Ycat, 256, wFp1[j], bFp1[j], nf, 256, Tfp);
    PNPP_GEMM(256, 2, Tfp, 256, wFp2[j], bFp2[j], nf, 128, xf_bufs[i]);
    cf = xf_bufs[i];
  }

  // ---- lin_out -----------------------------------------------------------
  PNPP_GEMM(128, 2, xfA, 128, wLo1, bLo1, N0, 128, Lout);
  {
    int groups = (N0 >> 4) * 1;          // Np = 16, NT = 1
    pnpp_gemm_wmma<128, 1, true><<<(groups + 7) / 8, 256, 0, stream>>>(
        Lout, 128, wLo2, bLo2, N0, 16,
        nullptr, out, /*ldc=*/2, /*nValid=*/2, /*relu=*/0);
  }
#undef PNPP_GEMM
}
